// MoE_52140902973543
// MI455X (gfx1250) — compile-verified, matrix operand order checked by reading, not run on hardware
//
#include <hip/hip_runtime.h>
#include <hip/hip_bf16.h>

// ---------------------------------------------------------------------------
// MoE: out[c] = sum_e softmax(x[c]@Wr^T)[e] * ( relu(x[c]@W1_e) @ W2_e ), c<1024
// B=4 S=2048 H=1024 I=4096 E=8, capacity=1024. Tokens >=1024 -> 0.
//
// GEMMs on v_wmma_f32_16x16x32_bf16 (wave32). Operands pre-arranged in
// fragment-native bf16 chunks:   chunk[lane][elem], 1KB each
//   lane = 16*((k>>3)&1) + (m_or_n & 15),  elem = (k&7) + 8*((k>>4)&1)
// so LDS tiles are linear copies of global chunks -> staged with
// global_load_async_to_lds_b128 (double-buffered, s_wait_asynccnt), and each
// WMMA fragment is one contiguous 32B LDS read (2x ds_load_b128).
// Block = 128 threads (4 waves), tile 128x128, per-wave 64x64 (16 WMMA/iter).
// GEMM2 is 4-way K-split into private fp32 partial buffers (deterministic).
// ---------------------------------------------------------------------------

typedef __attribute__((ext_vector_type(16))) __bf16 bf16x16;
typedef __attribute__((ext_vector_type(8)))  float  f32x8;
typedef unsigned short u16;

#define CAP   1024
#define Hdim  1024
#define Idim  4096
#define NEXP  8
#define NTOK  8192

#define BM 128            // block tile M (8 x 16-row chunks)
#define BN 128            // block tile N (8 x 16-col chunks)
#define TILE_BYTES 8192   // 128x32 bf16 == 32x128 bf16
#define BUF_BYTES  16384  // A tile + B tile
#define SMEM_BYTES 32768  // double buffered
#define KSPLIT 4          // GEMM2 K-split factor

// ---- CDNA5 async global->LDS copy: 64 bytes per thread (4 x b128) ----------
// ISA: INST_OFFSET is added to BOTH the LDS dest and the global src address.
__device__ __forceinline__ void async_copy64(unsigned lds_off, const void* sbase,
                                             unsigned goff)
{
    asm volatile("global_load_async_to_lds_b128 %0, %1, %2\n\t"
                 "global_load_async_to_lds_b128 %0, %1, %2 offset:16\n\t"
                 "global_load_async_to_lds_b128 %0, %1, %2 offset:32\n\t"
                 "global_load_async_to_lds_b128 %0, %1, %2 offset:48"
                 :: "v"(lds_off), "v"(goff), "s"(sbase)
                 : "memory");
}
__device__ __forceinline__ void wait_async0()
{
    asm volatile("s_wait_asynccnt 0x0" ::: "memory");
}

// ---------------------------------------------------------------------------
// WMMA mainloop. A chunks: [mchunk][kchunk], B chunks: [kchunk][nchunk].
// 4 waves: wave (wm 0..1, wn 0..1) owns 64x64 = 4x4 accumulators.
// ---------------------------------------------------------------------------
__device__ __forceinline__ void gemm_core(const u16* __restrict__ A, int kchunksA,
                                          const u16* __restrict__ B, int nchunksB,
                                          int mg0, int nb0, int kb0, int kcount,
                                          f32x8 acc[4][4])
{
    __shared__ __align__(32) unsigned char smem[SMEM_BYTES];
    const unsigned smem_base = (unsigned)(size_t)(void*)smem;

    const int tid  = threadIdx.x;       // 0..127
    const int lane = tid & 31;
    const int wid  = tid >> 5;          // 0..3
    const int wm   = wid & 1;
    const int wn   = wid >> 1;

    const int      ch    = tid >> 4;    // staging chunk 0..7 (16 thr/chunk)
    const unsigned inner = (tid & 15) * 64;

    // issue async copies for K-step `kb` into buffer (kb&1)
    auto issue = [&](int kb) {
        const unsigned bufo = (unsigned)(kb & 1) * BUF_BYTES;
        const unsigned gA =
            (unsigned)(((mg0 + ch) * kchunksA + kb0 + kb) * 1024) + inner;
        async_copy64(smem_base + bufo + ch * 1024 + inner, A, gA);
        const unsigned gB =
            (unsigned)(((kb0 + kb) * nchunksB + nb0) * 1024) + tid * 64;
        async_copy64(smem_base + bufo + TILE_BYTES + tid * 64, B, gB);
    };

    issue(0);
#pragma unroll 1
    for (int kb = 0; kb < kcount; ++kb) {
        wait_async0();           // this step's copies complete
        __syncthreads();         // everyone done reading the buffer we refill
        if (kb + 1 < kcount) issue(kb + 1);

        const unsigned bufo = (unsigned)(kb & 1) * BUF_BYTES;
        bf16x16 af[4], bfr[4];
#pragma unroll
        for (int mi = 0; mi < 4; ++mi)
            af[mi] = *(const bf16x16*)(smem + bufo +
                                       (unsigned)(wm * 4 + mi) * 1024 + lane * 32);
#pragma unroll
        for (int ni = 0; ni < 4; ++ni)
            bfr[ni] = *(const bf16x16*)(smem + bufo + TILE_BYTES +
                                        (unsigned)(wn * 4 + ni) * 1024 + lane * 32);
#pragma unroll
        for (int mi = 0; mi < 4; ++mi)
#pragma unroll
            for (int ni = 0; ni < 4; ++ni)
                acc[mi][ni] = __builtin_amdgcn_wmma_f32_16x16x32_bf16(
                    false, af[mi], false, bfr[ni], (short)0, acc[mi][ni],
                    false, false);
    }
}

__device__ __forceinline__ void zero_acc(f32x8 acc[4][4])
{
    const f32x8 z = {0.f, 0.f, 0.f, 0.f, 0.f, 0.f, 0.f, 0.f};
#pragma unroll
    for (int mi = 0; mi < 4; ++mi)
#pragma unroll
        for (int ni = 0; ni < 4; ++ni) acc[mi][ni] = z;
}

// ---------------------------------------------------------------------------
// GEMM1: H1' = relu( X'(1024x1024) @ W1'(1024x4096) ), written directly in
// GEMM2's A-fragment-native layout (K = Idim).
// ---------------------------------------------------------------------------
__global__ __launch_bounds__(128) void gemm1_kernel(const u16* __restrict__ Xf,
                                                    const u16* __restrict__ W1f,
                                                    u16* __restrict__ H1f)
{
    const int mg0 = blockIdx.y * (BM / 16);
    const int nb0 = blockIdx.x * (BN / 16);
    f32x8 acc[4][4];
    zero_acc(acc);
    gemm_core(Xf, Hdim / 32, W1f, Idim / 16, mg0, nb0, 0, Hdim / 32, acc);

    const int lane = threadIdx.x & 31;
    const int wid  = threadIdx.x >> 5;
    const int wm = wid & 1, wn = wid >> 1;
    const int half = lane >> 4, l16 = lane & 15;
    const int m0 = mg0 * 16, n0 = nb0 * 16;
#pragma unroll
    for (int mi = 0; mi < 4; ++mi)
#pragma unroll
        for (int ni = 0; ni < 4; ++ni) {
            const int i    = n0 + wn * 64 + ni * 16 + l16;   // K-dim of GEMM2
            const int ib   = i >> 5;
            const int h2   = (i >> 3) & 1;
            const int elem = (i & 7) + 8 * ((i >> 4) & 1);
#pragma unroll
            for (int r = 0; r < 8; ++r) {
                const int m = m0 + wm * 64 + mi * 16 + half * 8 + r;
                float v = acc[mi][ni][r];
                v = v > 0.f ? v : 0.f;                       // ReLU
                const size_t idx = ((size_t)(m >> 4) * (Idim / 32) + ib) * 512 +
                                   (size_t)(h2 * 16 + (m & 15)) * 16 + elem;
                H1f[idx] = __builtin_bit_cast(u16, (__bf16)v);
            }
        }
}

// ---------------------------------------------------------------------------
// GEMM2 (K-split): part[z] += w[c,e] * ( H1'[:, Kslice z] @ W2'[Kslice z, :] )
// ---------------------------------------------------------------------------
__global__ __launch_bounds__(128) void gemm2_kernel(const u16* __restrict__ H1f,
                                                    const u16* __restrict__ W2f,
                                                    const float* __restrict__ wgt,
                                                    int e, float* __restrict__ part)
{
    const int mg0 = blockIdx.y * (BM / 16);
    const int nb0 = blockIdx.x * (BN / 16);
    const int kcount = (Idim / 32) / KSPLIT;          // 32 chunks per slice
    const int kb0    = blockIdx.z * kcount;
    f32x8 acc[4][4];
    zero_acc(acc);
    gemm_core(H1f, Idim / 32, W2f, Hdim / 16, mg0, nb0, kb0, kcount, acc);

    float* p = part + (size_t)blockIdx.z * CAP * Hdim;
    const int lane = threadIdx.x & 31;
    const int wid  = threadIdx.x >> 5;
    const int wm = wid & 1, wn = wid >> 1;
    const int half = lane >> 4, l16 = lane & 15;
    const int m0 = mg0 * 16, n0 = nb0 * 16;
#pragma unroll
    for (int mi = 0; mi < 4; ++mi)
#pragma unroll
        for (int ni = 0; ni < 4; ++ni)
#pragma unroll
            for (int r = 0; r < 8; ++r) {
                const int m   = m0 + wm * 64 + mi * 16 + half * 8 + r;
                const int col = n0 + wn * 64 + ni * 16 + l16;
                const float scale = wgt[m * NEXP + e];
                p[(size_t)m * Hdim + col] += scale * acc[mi][ni][r];
            }
}

// ---------------------------------------------------------------------------
// Combine: out[m,:] = sum_z part[z][m,:]  (rows 0..1023)
// ---------------------------------------------------------------------------
__global__ __launch_bounds__(256) void combine_kernel(const float4* __restrict__ part,
                                                      float4* __restrict__ out)
{
    const size_t i = (size_t)blockIdx.x * blockDim.x + threadIdx.x; // 256K float4
    const size_t n4 = (size_t)CAP * Hdim / 4;
    float4 a = part[i];
    const float4 b = part[i + n4], c = part[i + 2 * n4], d = part[i + 3 * n4];
    a.x += b.x + c.x + d.x;
    a.y += b.y + c.y + d.y;
    a.z += b.z + c.z + d.z;
    a.w += b.w + c.w + d.w;
    out[i] = a;
}

// ---------------------------------------------------------------------------
// Weight conversion: fp32 row-major (K x N) -> bf16 fragment-native chunks
// dst[(k>>5)*(N/16) + (n>>4)][lane][elem]
// ---------------------------------------------------------------------------
__global__ __launch_bounds__(256) void convert_weight_kernel(const float* __restrict__ src,
                                                             u16* __restrict__ dst,
                                                             int K, int N)
{
    const size_t flat = ((size_t)blockIdx.x * blockDim.x + threadIdx.x) * 8;
    const int k = (int)(flat / N);
    const int n = (int)(flat % N);
    const float4* s = (const float4*)(src + (size_t)k * N + n);
    const float4 f0 = s[0], f1 = s[1];
    const int h    = (k >> 3) & 1;
    const int elem = (k & 7) + 8 * ((k >> 4) & 1);
    const size_t chunk = ((size_t)(k >> 5) * (N >> 4) + (n >> 4)) * 512;
    const int lbase = h * 16 + (n & 15);
    const float v[8] = {f0.x, f0.y, f0.z, f0.w, f1.x, f1.y, f1.z, f1.w};
#pragma unroll
    for (int j = 0; j < 8; ++j)
        dst[chunk + (size_t)(lbase + j) * 16 + elem] =
            __builtin_bit_cast(u16, (__bf16)v[j]);
}

// ---------------------------------------------------------------------------
// Router: softmax(x[c]@Wr^T + b) -> w[c,0..7]; x row -> bf16 fragment-native.
// One block per token, 256 threads x 4 elements.
// ---------------------------------------------------------------------------
__global__ __launch_bounds__(256) void router_kernel(const float* __restrict__ x,
                                                     const float* __restrict__ rw,
                                                     const float* __restrict__ rb,
                                                     float* __restrict__ wout,
                                                     u16* __restrict__ xf)
{
    const int c = blockIdx.x;
    const int t = threadIdx.x;
    __shared__ float red[256][NEXP];
    __shared__ float logits[NEXP];

    const float4 xv = ((const float4*)(x + (size_t)c * Hdim))[t];

    // fragment-native bf16 copy: k = 4t..4t+3 stay in one lane, 4 contig elems
    {
        const int kb    = t >> 3;
        const int h     = (t >> 1) & 1;
        const int elem0 = 4 * (t & 1) + 8 * ((t >> 2) & 1);
        u16* d = xf + ((size_t)(c >> 4) * (Hdim / 32) + kb) * 512 +
                 (size_t)(h * 16 + (c & 15)) * 16 + elem0;
        d[0] = __builtin_bit_cast(u16, (__bf16)xv.x);
        d[1] = __builtin_bit_cast(u16, (__bf16)xv.y);
        d[2] = __builtin_bit_cast(u16, (__bf16)xv.z);
        d[3] = __builtin_bit_cast(u16, (__bf16)xv.w);
    }
#pragma unroll
    for (int e = 0; e < NEXP; ++e) {
        const float4 wv = ((const float4*)(rw + (size_t)e * Hdim))[t];
        red[t][e] = xv.x * wv.x + xv.y * wv.y + xv.z * wv.z + xv.w * wv.w;
    }
    __syncthreads();
    if (t < NEXP) {
        float s = rb[t];
        for (int i = 0; i < 256; ++i) s += red[i][t];
        logits[t] = s;
    }
    __syncthreads();
    if (t == 0) {
        float mx = logits[0];
#pragma unroll
        for (int e = 1; e < NEXP; ++e) mx = fmaxf(mx, logits[e]);
        float ex[NEXP], s = 0.f;
#pragma unroll
        for (int e = 0; e < NEXP; ++e) { ex[e] = __expf(logits[e] - mx); s += ex[e]; }
        const float inv = 1.f / s;
#pragma unroll
        for (int e = 0; e < NEXP; ++e) wout[c * NEXP + e] = ex[e] * inv;
    }
}

__global__ void zero_kernel(float4* __restrict__ p, size_t n4)
{
    const size_t stride = (size_t)gridDim.x * blockDim.x;
    const float4 z = make_float4(0.f, 0.f, 0.f, 0.f);
    for (size_t i = (size_t)blockIdx.x * blockDim.x + threadIdx.x; i < n4; i += stride)
        p[i] = z;
}

// ---------------------------------------------------------------------------
extern "C" void kernel_launch(void* const* d_in, const int* in_sizes, int n_in,
                              void* d_out, int out_size, void* d_ws, size_t ws_size,
                              hipStream_t stream)
{
    const float* x  = (const float*)d_in[0];   // (B,S,H)
    const float* rw = (const float*)d_in[1];   // (E,H)
    const float* rb = (const float*)d_in[2];   // (E,)
    const float* w1 = (const float*)d_in[3];   // (E,H,I)
    const float* w2 = (const float*)d_in[4];   // (E,I,H)
    float* out = (float*)d_out;

    char* ws = (char*)d_ws;
    float* wgt  = (float*)ws;                                  // 32 KB
    u16*   xf   = (u16*)(ws + (64u << 10));                    // 2 MB  (X')
    u16*   h1f  = (u16*)(ws + (64u << 10) + (2u << 20));       // 8 MB  (H1')
    u16*   w1f  = (u16*)(ws + (64u << 10) + (10u << 20));      // 8 MB  (W1'_e)
    u16*   w2f  = (u16*)(ws + (64u << 10) + (18u << 20));      // 8 MB  (W2'_e)
    float* part = (float*)(ws + (64u << 10) + (26u << 20));    // 16 MB (K-split)

    zero_kernel<<<dim3(2048), 256, 0, stream>>>((float4*)out, (size_t)NTOK * Hdim / 4);
    zero_kernel<<<dim3(1024), 256, 0, stream>>>((float4*)part,
                                                (size_t)KSPLIT * CAP * Hdim / 4);
    router_kernel<<<dim3(CAP), 256, 0, stream>>>(x, rw, rb, wgt, xf);

    const int convB = (Hdim * Idim) / (256 * 8);   // 2048 blocks
    for (int e = 0; e < NEXP; ++e) {
        convert_weight_kernel<<<dim3(convB), 256, 0, stream>>>(
            w1 + (size_t)e * Hdim * Idim, w1f, Hdim, Idim);
        convert_weight_kernel<<<dim3(convB), 256, 0, stream>>>(
            w2 + (size_t)e * Idim * Hdim, w2f, Idim, Hdim);
        gemm1_kernel<<<dim3(Idim / BN, CAP / BM), 128, 0, stream>>>(xf, w1f, h1f);
        gemm2_kernel<<<dim3(Hdim / BN, CAP / BM, KSPLIT), 128, 0, stream>>>(
            h1f, w2f, wgt, e, part);
    }
    combine_kernel<<<dim3((CAP * Hdim / 4) / 256), 256, 0, stream>>>(
        (const float4*)part, (float4*)out);
}
